// RestoredImage_51960514347418
// MI455X (gfx1250) — matrix operand order repaired and checked by reading, not compiled, not run on hardware
//
#include <hip/hip_runtime.h>

typedef __attribute__((ext_vector_type(16))) _Float16 v16h;
typedef __attribute__((ext_vector_type(8)))  _Float16 v8h;
typedef __attribute__((ext_vector_type(8)))  float    v8f;

#define IMG_H 2160
#define IMG_W 3840
#define PAD   12
#define KS    25
#define KROW  40           // j range per dy row (16 outputs + 24 taps, rounded)
#define KTOT  1024         // 25*40 = 1000 useful K, padded to 32 chunks of 32
#define NCHUNK 32
#define TROWS 41           // staged rows per wave tile: m+dy reaches 15+25
#define TSTRIDE 1664       // per-wave staging region (41*40=1640, padded to mult of 128)
#define LAP_H 2182
#define LAP_W 3862
#define LAP_BLOCKS 1024

// ---------------------------------------------------------------------------
// 25x25 depthwise blur as implicit GEMM on v_wmma_f32_16x16x32_f16.
// K is packed linearly (k = dy*40 + j); staging rows are stored contiguously at
// stride 40 halfs so the A address is m*40 + k -> every lane's 8-half A segment
// is one aligned ds_load_b128, even across dy row boundaries.
// One block = 8 waves = 8 horizontally adjacent 16x16 output tiles of one channel.
// ---------------------------------------------------------------------------
__global__ __launch_bounds__(256) void blur25_wmma(const float* __restrict__ img,
                                                   const float* __restrict__ noise,
                                                   const float* __restrict__ g,
                                                   float* __restrict__ out)
{
    __shared__ v8h Bl8[(NCHUNK * 32 * 16) / 8];   // 16384 halfs: Toeplitz B in WMMA layout
    __shared__ v8h T8[(8 * TSTRIDE) / 8];         // 8 per-wave linear tiles, 13312 halfs
    _Float16* Bl = (_Float16*)Bl8;
    _Float16* T  = (_Float16*)T8;

    const int tid = threadIdx.x;
    const int ch  = blockIdx.z;
    const int y0  = blockIdx.y * 16;
    const int x0b = blockIdx.x * 128;

    // ---- Build B (Toeplitz of g) directly in WMMA B-register layout ----
    // b[i] of lane L is K = 32*chunk + (L&16) + i ; B[k,n] = g[k/40][k%40 - n] gated.
    for (int idx = tid; idx < NCHUNK * 32 * 16; idx += 256) {
        int i    = idx & 15;
        int lane = (idx >> 4) & 31;
        int cnk  = idx >> 9;
        int k    = 32 * cnk + (lane & 16) + i;
        int dy   = k / KROW;
        int q    = (k % KROW) - (lane & 15);
        float v  = (k < KS * KROW && q >= 0 && q < KS) ? g[dy * KS + q] : 0.0f;
        Bl[idx] = (_Float16)v;
    }

    // ---- Stage 8 per-wave tiles (41 rows x 40 cols, edge-clamped), fp32 -> fp16 ----
    const float* imgc = img + (size_t)ch * IMG_H * IMG_W;
    for (int t = tid; t < 8 * TROWS * KROW; t += 256) {
        int w   = t / (TROWS * KROW);
        int rem = t % (TROWS * KROW);
        int r   = rem / KROW, j = rem % KROW;
        int gy = y0 + r - PAD;            gy = gy < 0 ? 0 : (gy >= IMG_H ? IMG_H - 1 : gy);
        int gx = x0b + 16 * w + j - PAD;  gx = gx < 0 ? 0 : (gx >= IMG_W ? IMG_W - 1 : gx);
        T[w * TSTRIDE + r * KROW + j] = (_Float16)imgc[(size_t)gy * IMG_W + gx];
    }
    __syncthreads();

    // ---- Per-wave 16x16 tile: 32 chained WMMAs ----
    const int wv    = tid >> 5;
    const int lane  = tid & 31;
    const int m     = lane & 15;           // A-matrix M row / C-matrix N col index
    const int hi    = lane >> 4;
    const int koffA = hi ? 16 : 0;         // A layout K offset for this lane group
    const int abase = wv * TSTRIDE + m * KROW + koffA;   // halfs; 16B aligned

    v8f c = {};
    #pragma unroll
    for (int cnk = 0; cnk < NCHUNK; ++cnk) {
        const v8h* ap = (const v8h*)&T[abase + 32 * cnk];
        v8h alo = ap[0];                                // K = 32c+koffA .. +7
        v8h ahi = ap[1];                                // K = 32c+8+koffA .. +7
        v16h A = __builtin_shufflevector(alo, ahi, 0,1,2,3,4,5,6,7,8,9,10,11,12,13,14,15);

        const v8h* bp = Bl8 + cnk * 64 + lane * 2;
        v8h blo = bp[0];
        v8h bhi = bp[1];
        v16h B = __builtin_shufflevector(blo, bhi, 0,1,2,3,4,5,6,7,8,9,10,11,12,13,14,15);

        c = __builtin_amdgcn_wmma_f32_16x16x32_f16(false, A, false, B,
                                                   (short)0, c, false, false);
    }

    // ---- Store C (+ noise broadcast over channels) ----
    const int xo    = x0b + 16 * wv + m;   // C layout: lane&15 is the N column
    const int mbase = hi ? 8 : 0;
    float* outc = out + (size_t)ch * IMG_H * IMG_W;
    #pragma unroll
    for (int v = 0; v < 8; ++v) {
        int yo = y0 + mbase + v;
        size_t off = (size_t)yo * IMG_W + xo;
        outc[off] = c[v] + noise[off];
    }
}

// ---------------------------------------------------------------------------
// Laplacian regularizer: mean over (3, 2182, 3862) of (4*p - up - down - l - r)^2
// on the edge-replicated padded image. Deterministic two-pass reduction.
// ---------------------------------------------------------------------------
__device__ __forceinline__ float padv(const float* __restrict__ imgc, int r, int cc) {
    int y = r  - PAD; y = y < 0 ? 0 : (y >= IMG_H ? IMG_H - 1 : y);
    int x = cc - PAD; x = x < 0 ? 0 : (x >= IMG_W ? IMG_W - 1 : x);
    return imgc[(size_t)y * IMG_W + x];
}

__global__ __launch_bounds__(256) void lap_partial(const float* __restrict__ img,
                                                   float* __restrict__ ws)
{
    __shared__ float red[256];
    const long long HW = (long long)LAP_H * LAP_W;
    const long long N2 = 3LL * HW;
    float s = 0.0f;
    for (long long i = (long long)blockIdx.x * 256 + threadIdx.x; i < N2;
         i += (long long)gridDim.x * 256) {
        int c   = (int)(i / HW);
        int rem = (int)(i % HW);
        int y   = rem / LAP_W;
        int x   = rem % LAP_W;
        const float* imgc = img + (size_t)c * IMG_H * IMG_W;
        float v = 4.0f * padv(imgc, y + 1, x + 1)
                - padv(imgc, y,     x + 1)
                - padv(imgc, y + 2, x + 1)
                - padv(imgc, y + 1, x)
                - padv(imgc, y + 1, x + 2);
        s += v * v;
    }
    red[threadIdx.x] = s;
    __syncthreads();
    for (int off = 128; off; off >>= 1) {
        if (threadIdx.x < off) red[threadIdx.x] += red[threadIdx.x + off];
        __syncthreads();
    }
    if (threadIdx.x == 0) ws[blockIdx.x] = red[0];
}

__global__ __launch_bounds__(256) void lap_final(const float* __restrict__ ws,
                                                 float* __restrict__ outscalar)
{
    __shared__ float red[256];
    float s = ws[threadIdx.x] + ws[threadIdx.x + 256]
            + ws[threadIdx.x + 512] + ws[threadIdx.x + 768];
    red[threadIdx.x] = s;
    __syncthreads();
    for (int off = 128; off; off >>= 1) {
        if (threadIdx.x < off) red[threadIdx.x] += red[threadIdx.x + off];
        __syncthreads();
    }
    if (threadIdx.x == 0)
        *outscalar = red[0] / (float)(3LL * LAP_H * LAP_W);
}

// ---------------------------------------------------------------------------
extern "C" void kernel_launch(void* const* d_in, const int* in_sizes, int n_in,
                              void* d_out, int out_size, void* d_ws, size_t ws_size,
                              hipStream_t stream)
{
    const float* img   = (const float*)d_in[0];   // (1,3,2160,3840) f32
    const float* noise = (const float*)d_in[1];   // (1,1,2160,3840) f32
    const float* g     = (const float*)d_in[2];   // (25,25) f32
    float* out = (float*)d_out;
    float* ws  = (float*)d_ws;                    // needs LAP_BLOCKS*4 bytes

    // 3840/128 = 30 tile-groups of 8 waves, 2160/16 = 135 tile rows, 3 channels
    dim3 grid(IMG_W / 128, IMG_H / 16, 3);
    blur25_wmma<<<grid, 256, 0, stream>>>(img, noise, g, out);

    lap_partial<<<LAP_BLOCKS, 256, 0, stream>>>(img, ws);
    lap_final<<<1, 256, 0, stream>>>(ws, out + (out_size - 1));
}
